// TransformerBlockWithSSM_64201171140928
// MI455X (gfx1250) — compile-verified, matrix operand order checked by reading, not run on hardware
//
#include <hip/hip_runtime.h>

// ---------------------------------------------------------------------------
// Problem dimensions (match reference)
// ---------------------------------------------------------------------------
#define BATCH   2
#define SEQ     2048
#define DMODEL  1024
#define DINNER  2048
#define NSTATE  64
#define DTRANK  64
#define HIDDEN  4096
#define MROWS   (BATCH * SEQ)      // 4096 token rows

#define LDS_PAD 40                 // bf16 elements per staged row (32 + 8 pad)
#define TILE_E  (128 * LDS_PAD)    // elements per 128x32 staged tile

typedef __attribute__((ext_vector_type(16))) __bf16          v16bf;
typedef __attribute__((ext_vector_type(16))) unsigned short  v16us;
typedef __attribute__((ext_vector_type(8)))  unsigned short  v8us;
typedef __attribute__((ext_vector_type(8)))  float           v8f;
typedef unsigned short ushort_t;

// f32 -> bf16 round-to-nearest-even (epilogues / convert passes only).
__device__ __forceinline__ ushort_t f2bf(float f) {
    unsigned int u = __float_as_uint(f);
    u += 0x7FFFu + ((u >> 16) & 1u);
    return (ushort_t)(u >> 16);
}

// ---------------------------------------------------------------------------
// WMMA bf16 fragment load (CDNA5 ISA 7.12.2, wave32): lane l needs elements
// {k..k+7} and {k+16..k+23} of its row; p pre-offset by khalf*8.
// Two 16-byte loads, zero ALU. Works for global and LDS pointers.
// ---------------------------------------------------------------------------
__device__ __forceinline__ v16bf load_frag_bf(const ushort_t* __restrict__ p) {
    v8us lo = *(const v8us*)p;
    v8us hi = *(const v8us*)(p + 16);
    v16us t = __builtin_shufflevector(lo, hi, 0, 1, 2, 3, 4, 5, 6, 7,
                                      8, 9, 10, 11, 12, 13, 14, 15);
    return __builtin_bit_cast(v16bf, t);
}

// Shared epilogue scalar op.
template <int EPI>
__device__ __forceinline__ float epi_op(float v, float bv, float rv) {
    v += bv;
    if (EPI == 1) v = (v > 20.0f) ? v : log1pf(expf(v));            // softplus
    if (EPI == 2) v = 0.5f * v * (1.0f + erff(v * 0.70710678118f)); // gelu
    if (EPI == 3 || EPI == 4) v += rv;
    return v;
}

// ---------------------------------------------------------------------------
// Register-tile WMMA GEMM (for odd shapes: N=192 / K=64).
// One wave -> 16(M) x 64(N). C = epi(A[M,K]bf16 @ W[N,K]bf16^T)
// ---------------------------------------------------------------------------
template <int EPI, bool OUTBF>
__global__ void wmma_gemm_reg_kernel(const ushort_t* __restrict__ A,
                                     const ushort_t* __restrict__ W,
                                     void* __restrict__ Cout,
                                     int M, int N, int K, int lda,
                                     const float* __restrict__ bias,
                                     const float* __restrict__ res) {
    int gw      = (blockIdx.x * blockDim.x + threadIdx.x) >> 5;
    int tilesN  = N >> 6;
    int totalW  = (M >> 4) * tilesN;
    if (gw >= totalW) return;
    int tm    = gw / tilesN;
    int tn    = gw - tm * tilesN;
    int lane  = threadIdx.x & 31;
    int lh    = lane & 15;
    int khalf = (lane >> 4) & 1;

    const ushort_t* arow  = A + (size_t)(tm * 16 + lh) * lda + khalf * 8;
    const ushort_t* wrow0 = W + (size_t)(tn * 64 + lh) * K + khalf * 8;

    v8f acc[4] = {};
    for (int k = 0; k < K; k += 32) {
        v16bf af = load_frag_bf(arow + k);
#pragma unroll
        for (int s = 0; s < 4; ++s) {
            v16bf bf = load_frag_bf(wrow0 + (size_t)s * 16 * K + k);
            acc[s] = __builtin_amdgcn_wmma_f32_16x16x32_bf16(
                false, af, false, bf, (short)0, acc[s], false, false);
        }
    }

    int rbase = tm * 16 + khalf * 8;
    float*    Cf = (float*)Cout;
    ushort_t* Cb = (ushort_t*)Cout;
#pragma unroll
    for (int s = 0; s < 4; ++s) {
        int col  = tn * 64 + s * 16 + lh;
        float bv = (EPI == 1 || EPI == 2 || EPI == 4) ? bias[col] : 0.0f;
#pragma unroll
        for (int j = 0; j < 8; ++j) {
            size_t idx = (size_t)(rbase + j) * N + col;
            float rv = (EPI == 3 || EPI == 4) ? res[idx] : 0.0f;
            float v  = epi_op<EPI>(acc[s][j], bv, rv);
            if (OUTBF) Cb[idx] = f2bf(v);
            else       Cf[idx] = v;
        }
    }
}

// ---------------------------------------------------------------------------
// Double-buffered LDS-tiled WMMA GEMM for the big GEMMs (M%128==0,
// N%128==0, K%32==0, lda==K). Block = 256 threads = 8 waves as 4(M)x2(N);
// block tile 128x128; K staged 32/step through ping-pong LDS buffers
// (2 x (10KB A + 10KB B), rows padded to 40 bf16 -> conflict-free
// ds_load_b128). One barrier per K-step: ds_stores of tile i+1 overlap the
// 8 WMMAs of tile i; global loads issued before compute.
// ---------------------------------------------------------------------------
template <int EPI, bool OUTBF>
__global__ __launch_bounds__(256)
void wmma_gemm_lds_kernel(const ushort_t* __restrict__ A,
                          const ushort_t* __restrict__ W,
                          void* __restrict__ Cout,
                          int M, int N, int K,
                          const float* __restrict__ bias,
                          const float* __restrict__ res) {
    __shared__ ushort_t sA[2 * TILE_E];
    __shared__ ushort_t sB[2 * TILE_E];

    int blocksN = N >> 7;
    int bm = blockIdx.x / blocksN;
    int bn = blockIdx.x - bm * blocksN;

    int tid   = threadIdx.x;
    int wid   = tid >> 5;
    int wm    = wid & 3;          // 0..3  -> M offset wm*32
    int wn    = wid >> 2;         // 0..1  -> N offset wn*64
    int lane  = tid & 31;
    int lh    = lane & 15;
    int khalf = (lane >> 4) & 1;

    // Staging: 128x32 bf16 tile = 4096 elems; 256 threads x 8 elems x 2 iters.
    int e0 = tid * 8;
    int r0 = e0 >> 5, c0 = e0 & 31;               // iter0: rows 0..63
    const ushort_t* gA0 = A + (size_t)(bm * 128 + r0) * K + c0;
    const ushort_t* gA1 = gA0 + (size_t)64 * K;   // iter1: rows 64..127
    const ushort_t* gB0 = W + (size_t)(bn * 128 + r0) * K + c0;
    const ushort_t* gB1 = gB0 + (size_t)64 * K;
    int sOff0 = r0 * LDS_PAD + c0;
    int sOff1 = sOff0 + 64 * LDS_PAD;

    // Compute-side LDS offsets (pre-offset by khalf*8).
    int oA0 = (wm * 32 + lh) * LDS_PAD + khalf * 8;
    int oA1 = oA0 + 16 * LDS_PAD;
    int oB  = (wn * 64 + lh) * LDS_PAD + khalf * 8;

    int nt = K >> 5;              // #K-steps (>= 2 for all call sites)
    v8f acc[2][4] = {};

    // --- prologue: stage tile 0 into buffer 0 ---
    {
        v8us ra0 = *(const v8us*)gA0;
        v8us ra1 = *(const v8us*)gA1;
        v8us rb0 = *(const v8us*)gB0;
        v8us rb1 = *(const v8us*)gB1;
        *(v8us*)(sA + sOff0) = ra0;
        *(v8us*)(sA + sOff1) = ra1;
        *(v8us*)(sB + sOff0) = rb0;
        *(v8us*)(sB + sOff1) = rb1;
    }
    __syncthreads();

#pragma unroll 2
    for (int i = 0; i < nt - 1; ++i) {
        int p  = i & 1;
        int k1 = (i + 1) << 5;
        // issue global loads for tile i+1 (fill during WMMAs below)
        v8us ra0 = *(const v8us*)(gA0 + k1);
        v8us ra1 = *(const v8us*)(gA1 + k1);
        v8us rb0 = *(const v8us*)(gB0 + k1);
        v8us rb1 = *(const v8us*)(gB1 + k1);

        // compute on buffer p
        const ushort_t* bA = sA + p * TILE_E;
        const ushort_t* bB = sB + p * TILE_E;
        v16bf a0 = load_frag_bf(bA + oA0);
        v16bf a1 = load_frag_bf(bA + oA1);
#pragma unroll
        for (int s = 0; s < 4; ++s) {
            v16bf bf = load_frag_bf(bB + oB + s * 16 * LDS_PAD);
            acc[0][s] = __builtin_amdgcn_wmma_f32_16x16x32_bf16(
                false, a0, false, bf, (short)0, acc[0][s], false, false);
            acc[1][s] = __builtin_amdgcn_wmma_f32_16x16x32_bf16(
                false, a1, false, bf, (short)0, acc[1][s], false, false);
        }

        // store tile i+1 into the other buffer (overlaps WMMAs above)
        int q = p ^ 1;
        *(v8us*)(sA + q * TILE_E + sOff0) = ra0;
        *(v8us*)(sA + q * TILE_E + sOff1) = ra1;
        *(v8us*)(sB + q * TILE_E + sOff0) = rb0;
        *(v8us*)(sB + q * TILE_E + sOff1) = rb1;
        __syncthreads();   // publish tile i+1; fence reads of buffer p
    }

    // --- epilogue compute on last tile ---
    {
        int p = (nt - 1) & 1;
        const ushort_t* bA = sA + p * TILE_E;
        const ushort_t* bB = sB + p * TILE_E;
        v16bf a0 = load_frag_bf(bA + oA0);
        v16bf a1 = load_frag_bf(bA + oA1);
#pragma unroll
        for (int s = 0; s < 4; ++s) {
            v16bf bf = load_frag_bf(bB + oB + s * 16 * LDS_PAD);
            acc[0][s] = __builtin_amdgcn_wmma_f32_16x16x32_bf16(
                false, a0, false, bf, (short)0, acc[0][s], false, false);
            acc[1][s] = __builtin_amdgcn_wmma_f32_16x16x32_bf16(
                false, a1, false, bf, (short)0, acc[1][s], false, false);
        }
    }

    float*    Cf = (float*)Cout;
    ushort_t* Cb = (ushort_t*)Cout;
#pragma unroll
    for (int fi = 0; fi < 2; ++fi) {
        int rbase = bm * 128 + wm * 32 + fi * 16 + khalf * 8;
#pragma unroll
        for (int s = 0; s < 4; ++s) {
            int col  = bn * 128 + wn * 64 + s * 16 + lh;
            float bv = (EPI == 1 || EPI == 2 || EPI == 4) ? bias[col] : 0.0f;
#pragma unroll
            for (int j = 0; j < 8; ++j) {
                size_t idx = (size_t)(rbase + j) * N + col;
                float rv = (EPI == 3 || EPI == 4) ? res[idx] : 0.0f;
                float v  = epi_op<EPI>(acc[fi][s][j], bv, rv);
                if (OUTBF) Cb[idx] = f2bf(v);
                else       Cf[idx] = v;
            }
        }
    }
}

// ---------------------------------------------------------------------------
// Bulk f32 -> bf16 convert (8 elements / thread, n multiple of 8)
// ---------------------------------------------------------------------------
__global__ void cvt_bf16_kernel(const float* __restrict__ in,
                                ushort_t* __restrict__ out, long n) {
    long i = ((long)blockIdx.x * blockDim.x + threadIdx.x) * 8;
    if (i >= n) return;
    const float4* p = (const float4*)(in + i);
    float4 a = p[0], b = p[1];
    v8us o;
    o[0] = f2bf(a.x); o[1] = f2bf(a.y); o[2] = f2bf(a.z); o[3] = f2bf(a.w);
    o[4] = f2bf(b.x); o[5] = f2bf(b.y); o[6] = f2bf(b.z); o[7] = f2bf(b.w);
    *(v8us*)(out + i) = o;
}

// Extract dt columns (0..63) of x_dbl (ld=192) into contiguous bf16 (ld=64)
__global__ void extract_dt_kernel(const float* __restrict__ x_dbl,
                                  ushort_t* __restrict__ dt_bf) {
    int idx = blockIdx.x * blockDim.x + threadIdx.x;
    if (idx >= MROWS * DTRANK) return;
    int r = idx >> 6, c = idx & 63;
    dt_bf[idx] = f2bf(x_dbl[(size_t)r * (DTRANK + 2 * NSTATE) + c]);
}

// ---------------------------------------------------------------------------
// LayerNorm over rows (one block per row). BF16OUT writes bf16.
// ---------------------------------------------------------------------------
template <bool BF16OUT>
__global__ void ln_kernel(const float* __restrict__ in,
                          const float* __restrict__ w,
                          const float* __restrict__ b,
                          void* __restrict__ out, int D) {
    __shared__ float s_sum[8], s_sq[8];
    int row = blockIdx.x;
    const float* xr = in + (size_t)row * D;
    float s = 0.0f, ss = 0.0f;
    for (int i = threadIdx.x; i < D; i += blockDim.x) {
        float v = xr[i];
        s += v; ss += v * v;
    }
    for (int off = 16; off > 0; off >>= 1) {
        s  += __shfl_xor(s, off, 32);
        ss += __shfl_xor(ss, off, 32);
    }
    int wv = threadIdx.x >> 5, ln = threadIdx.x & 31;
    if (ln == 0) { s_sum[wv] = s; s_sq[wv] = ss; }
    __syncthreads();
    if (wv == 0) {
        s  = (ln < 8) ? s_sum[ln] : 0.0f;
        ss = (ln < 8) ? s_sq[ln]  : 0.0f;
        for (int off = 4; off > 0; off >>= 1) {
            s  += __shfl_xor(s, off, 32);
            ss += __shfl_xor(ss, off, 32);
        }
        if (ln == 0) { s_sum[0] = s; s_sq[0] = ss; }
    }
    __syncthreads();
    float mu  = s_sum[0] / (float)D;
    float var = s_sq[0] / (float)D - mu * mu;
    float inv = rsqrtf(var + 1e-5f);
    for (int i = threadIdx.x; i < D; i += blockDim.x) {
        float v = (xr[i] - mu) * inv * w[i] + b[i];
        if (BF16OUT) ((ushort_t*)out)[(size_t)row * D + i] = f2bf(v);
        else         ((float*)out)[(size_t)row * D + i] = v;
    }
}

// ---------------------------------------------------------------------------
// Depthwise causal conv1d (kernel 4) + SiLU -> f32 (scan u) + bf16 (GEMM A)
// ---------------------------------------------------------------------------
__global__ void conv_silu_kernel(const float* __restrict__ xz,
                                 const float* __restrict__ cw,
                                 const float* __restrict__ cb,
                                 float* __restrict__ out_f,
                                 ushort_t* __restrict__ out_bf) {
    int idx = blockIdx.x * blockDim.x + threadIdx.x;
    if (idx >= BATCH * SEQ * DINNER) return;
    int c = idx & (DINNER - 1);
    int l = (idx >> 11) & (SEQ - 1);
    int b = idx / (SEQ * DINNER);
    float acc = cb[c];
#pragma unroll
    for (int j = 0; j < 4; ++j) {
        int li = l - 3 + j;
        if (li >= 0)
            acc += cw[c * 4 + j] *
                   xz[(size_t)(b * SEQ + li) * (2 * DINNER) + c];
    }
    float s = acc / (1.0f + expf(-acc));   // silu
    out_f[idx]  = s;
    out_bf[idx] = f2bf(s);
}

// ---------------------------------------------------------------------------
// Selective scan: one wave per (b, d); lane owns states n=lane, n=lane+32;
// shuffle-reduce; fused u*D + silu(z) gate; bf16 output.
// ---------------------------------------------------------------------------
__global__ void scan_kernel(const float* __restrict__ u,
                            const float* __restrict__ delta,
                            const float* __restrict__ x_dbl,
                            const float* __restrict__ xz,
                            const float* __restrict__ A_log,
                            const float* __restrict__ Dp,
                            ushort_t* __restrict__ y_out) {
    int wid  = (blockIdx.x * blockDim.x + threadIdx.x) >> 5;
    int lane = threadIdx.x & 31;
    if (wid >= BATCH * DINNER) return;
    int b = wid / DINNER;
    int d = wid - b * DINNER;
    int n0 = lane, n1 = lane + 32;

    float A0 = -expf(A_log[(size_t)d * NSTATE + n0]);
    float A1 = -expf(A_log[(size_t)d * NSTATE + n1]);
    float Dv = Dp[d];
    float h0 = 0.0f, h1 = 0.0f;

    for (int t = 0; t < SEQ; ++t) {
        size_t row = (size_t)(b * SEQ + t);
        float dt = delta[row * DINNER + d];
        float uv = u[row * DINNER + d];
        const float* bc = x_dbl + row * (DTRANK + 2 * NSTATE);
        float Bn0 = bc[DTRANK + n0];
        float Bn1 = bc[DTRANK + n1];
        float Cn0 = bc[DTRANK + NSTATE + n0];
        float Cn1 = bc[DTRANK + NSTATE + n1];
        h0 = expf(dt * A0) * h0 + dt * Bn0 * uv;
        h1 = expf(dt * A1) * h1 + dt * Bn1 * uv;
        float p = h0 * Cn0 + h1 * Cn1;
        for (int off = 16; off > 0; off >>= 1) p += __shfl_xor(p, off, 32);
        if (lane == 0) {
            float z   = xz[row * (2 * DINNER) + DINNER + d];
            float sil = z / (1.0f + expf(-z));
            y_out[row * DINNER + d] = f2bf((p + uv * Dv) * sil);
        }
    }
}

// ---------------------------------------------------------------------------
// Host-side launchers
// ---------------------------------------------------------------------------
template <int EPI, bool OUTBF>
static void launch_gemm_reg(const ushort_t* A, const ushort_t* W, void* C,
                            int M, int N, int K, int lda,
                            const float* bias, const float* res,
                            hipStream_t stream) {
    int waves  = (M / 16) * (N / 64);
    int blocks = (waves + 7) / 8;
    wmma_gemm_reg_kernel<EPI, OUTBF><<<blocks, 256, 0, stream>>>(
        A, W, C, M, N, K, lda, bias, res);
}

template <int EPI, bool OUTBF>
static void launch_gemm_lds(const ushort_t* A, const ushort_t* W, void* C,
                            int M, int N, int K,
                            const float* bias, const float* res,
                            hipStream_t stream) {
    int blocks = (M / 128) * (N / 128);
    wmma_gemm_lds_kernel<EPI, OUTBF><<<blocks, 256, 0, stream>>>(
        A, W, C, M, N, K, bias, res);
}

static void launch_cvt(const float* in, ushort_t* out, long n,
                       hipStream_t stream) {
    long th = n / 8;
    cvt_bf16_kernel<<<(int)((th + 255) / 256), 256, 0, stream>>>(in, out, n);
}

extern "C" void kernel_launch(void* const* d_in, const int* in_sizes, int n_in,
                              void* d_out, int out_size, void* d_ws,
                              size_t ws_size, hipStream_t stream) {
    const float* x         = (const float*)d_in[0];
    const float* ln1_w     = (const float*)d_in[1];
    const float* ln1_b     = (const float*)d_in[2];
    const float* ln4_w     = (const float*)d_in[3];
    const float* ln4_b     = (const float*)d_in[4];
    const float* ln5_w     = (const float*)d_in[5];
    const float* ln5_b     = (const float*)d_in[6];
    const float* ln6_w     = (const float*)d_in[7];
    const float* ln6_b     = (const float*)d_in[8];
    const float* in_proj_w = (const float*)d_in[9];
    const float* conv_w    = (const float*)d_in[10];
    const float* conv_b    = (const float*)d_in[11];
    const float* x_proj_w  = (const float*)d_in[12];
    const float* dt_proj_w = (const float*)d_in[13];
    const float* dt_proj_b = (const float*)d_in[14];
    const float* A_log     = (const float*)d_in[15];
    const float* Dp        = (const float*)d_in[16];
    const float* out_proj_w= (const float*)d_in[17];
    const float* mlp_w1    = (const float*)d_in[18];
    const float* mlp_b1    = (const float*)d_in[19];
    const float* mlp_w2    = (const float*)d_in[20];
    const float* mlp_b2    = (const float*)d_in[21];

    // ---- workspace layout (256B-aligned bump allocator) ----
    char* base = (char*)d_ws;
    size_t o = 0;
    auto alloc = [&](size_t bytes) -> char* {
        char* p = base + o;
        o += (bytes + 255) & ~(size_t)255;
        return p;
    };
    float* xz      = (float*)alloc((size_t)MROWS * 2 * DINNER * 4);  // aliased by h1_bf later
    float* x_dbl   = (float*)alloc((size_t)MROWS * (DTRANK + 2 * NSTATE) * 4);
    float* delta   = (float*)alloc((size_t)MROWS * DINNER * 4);
    float* u_conv  = (float*)alloc((size_t)MROWS * DINNER * 4);      // aliased by x6 later
    float* x_res   = (float*)alloc((size_t)MROWS * DMODEL * 4);
    float* x4      = (float*)alloc((size_t)MROWS * DMODEL * 4);
    ushort_t* xn_bf  = (ushort_t*)alloc((size_t)MROWS * DMODEL * 2);
    ushort_t* xi_bf  = (ushort_t*)alloc((size_t)MROWS * DINNER * 2);
    ushort_t* dt_bf  = (ushort_t*)alloc((size_t)MROWS * DTRANK * 2);
    ushort_t* y_bf   = (ushort_t*)alloc((size_t)MROWS * DINNER * 2);
    ushort_t* h5_bf  = (ushort_t*)alloc((size_t)MROWS * DMODEL * 2);
    ushort_t* w_in   = (ushort_t*)alloc((size_t)2 * DINNER * DMODEL * 2);
    ushort_t* w_xp   = (ushort_t*)alloc((size_t)(DTRANK + 2 * NSTATE) * DINNER * 2);
    ushort_t* w_dt   = (ushort_t*)alloc((size_t)DINNER * DTRANK * 2);
    ushort_t* w_out  = (ushort_t*)alloc((size_t)DMODEL * DINNER * 2);
    ushort_t* w_m1   = (ushort_t*)alloc((size_t)HIDDEN * DMODEL * 2);
    ushort_t* w_m2   = (ushort_t*)alloc((size_t)DMODEL * HIDDEN * 2);
    ushort_t* h1_bf  = (ushort_t*)xz;     // xz free after scan (z-gate consumed)
    float*    x6     = u_conv;            // u_conv free after scan

    // ---- weight conversions (one pass each; bandwidth-trivial) ----
    launch_cvt(in_proj_w, w_in, (long)2 * DINNER * DMODEL, stream);
    launch_cvt(x_proj_w,  w_xp, (long)(DTRANK + 2 * NSTATE) * DINNER, stream);
    launch_cvt(dt_proj_w, w_dt, (long)DINNER * DTRANK, stream);
    launch_cvt(out_proj_w,w_out,(long)DMODEL * DINNER, stream);
    launch_cvt(mlp_w1,    w_m1, (long)HIDDEN * DMODEL, stream);
    launch_cvt(mlp_w2,    w_m2, (long)DMODEL * HIDDEN, stream);

    // 1) xn_bf = LN1(x)
    ln_kernel<true><<<MROWS, 256, 0, stream>>>(x, ln1_w, ln1_b, xn_bf, DMODEL);
    // 2) xz = xn @ in_proj_w^T           (4096 x 4096 x 1024)  [LDS-tiled]
    launch_gemm_lds<0, false>(xn_bf, w_in, xz, MROWS, 2 * DINNER, DMODEL,
                              nullptr, nullptr, stream);
    // 3) u_conv / xi_bf = silu(causal_dwconv(xi) + conv_b)
    {
        int tot = BATCH * SEQ * DINNER;
        conv_silu_kernel<<<(tot + 255) / 256, 256, 0, stream>>>(
            xz, conv_w, conv_b, u_conv, xi_bf);
    }
    // 4) x_dbl = xi @ x_proj_w^T         (4096 x 192 x 2048)   [reg tile]
    launch_gemm_reg<0, false>(xi_bf, w_xp, x_dbl, MROWS, DTRANK + 2 * NSTATE,
                              DINNER, DINNER, nullptr, nullptr, stream);
    // 5) dt_bf = bf16(x_dbl[:, :64])
    extract_dt_kernel<<<(MROWS * DTRANK + 255) / 256, 256, 0, stream>>>(
        x_dbl, dt_bf);
    // 6) delta = softplus(dt @ dt_proj_w^T + dt_proj_b)  (K=64) [reg tile]
    launch_gemm_reg<1, false>(dt_bf, w_dt, delta, MROWS, DINNER, DTRANK,
                              DTRANK, dt_proj_b, nullptr, stream);
    // 7) selective scan + u*D + silu(z) gate -> y_bf
    {
        int waves  = BATCH * DINNER;
        int blocks = (waves * 32 + 255) / 256;
        scan_kernel<<<blocks, 256, 0, stream>>>(u_conv, delta, x_dbl, xz,
                                                A_log, Dp, y_bf);
    }
    // 8) x_res = x + y @ out_proj_w^T    (4096 x 1024 x 2048)  [LDS-tiled]
    launch_gemm_lds<3, false>(y_bf, w_out, x_res, MROWS, DMODEL, DINNER,
                              nullptr, x, stream);
    // 9) x4 = LN4(x_res)
    ln_kernel<false><<<MROWS, 256, 0, stream>>>(x_res, ln4_w, ln4_b, x4, DMODEL);
    // 10) h5_bf = LN5(x4)
    ln_kernel<true><<<MROWS, 256, 0, stream>>>(x4, ln5_w, ln5_b, h5_bf, DMODEL);
    // 11) h1_bf = gelu(h5 @ mlp_w1^T + mlp_b1)  (4096x4096x1024) [LDS-tiled]
    launch_gemm_lds<2, true>(h5_bf, w_m1, h1_bf, MROWS, HIDDEN, DMODEL,
                             mlp_b1, nullptr, stream);
    // 12) x6 = x4 + h1 @ mlp_w2^T + mlp_b2      (4096x1024x4096) [LDS-tiled]
    launch_gemm_lds<4, false>(h1_bf, w_m2, x6, MROWS, DMODEL, HIDDEN,
                              mlp_b2, x4, stream);
    // 13) out = LN6(x6)
    ln_kernel<false><<<MROWS, 256, 0, stream>>>(x6, ln6_w, ln6_b, d_out, DMODEL);

    (void)in_sizes; (void)n_in; (void)out_size; (void)ws_size;
}
